// TokenLevelRouter_50964081934534
// MI455X (gfx1250) — matrix operand order repaired and checked by reading, not compiled, not run on hardware
//
#include <hip/hip_runtime.h>
#include <hip/hip_bf16.h>

// ---------------------------------------------------------------------------
// TokenLevelRouter for MI455X (gfx1250, wave32, WMMA).
//
// Math reduction: gate = sigmoid(..) is a strictly positive per-token scalar
// and softmax is monotone, so argmax(softmax(scores*gate)) == argmax(scores).
// The whole gate network (Wg1/bg1/Wg2/bg2) and the softmax are dead code for
// the output. We only need:
//   idx = argmax_e( relu(X @ W1 + b1) @ W2 + b2 )
//   out = X * expert_scales[idx] + expert_biases[idx]
// GEMM1 (8192x1024x512) runs on v_wmma_f32_16x16x32_bf16 with f32 accum.
// W1 K-chunks are double-buffered in LDS; fp32->bf16 packing of the A
// fragments uses a single v_perm_b32 per pair.
// ---------------------------------------------------------------------------

typedef __attribute__((ext_vector_type(16))) __bf16 v16bf;
typedef __attribute__((ext_vector_type(8)))  float  v8f;

#define H   1024
#define RH  512
#define E   16
#define KC  32          // K-chunk (one wmma K)
#define NKC (H / KC)    // 32 chunks
#define TOKS_PER_BLOCK 64
#define THREADS 256
#define WCOL_STRIDE 40  // padded halves per staged W1 column (80B, 16B aligned,
                        // stride of 20 banks -> 16 lanes hit 16 distinct banks)
#define WBUF (RH * WCOL_STRIDE)   // halves per staging buffer
#define HSTRIDE 520     // padded halves per hidden row

__device__ __forceinline__ unsigned short f32_bf16(float f) {
    unsigned u = __float_as_uint(f);
    u += 0x7FFFu + ((u >> 16) & 1u);           // round-to-nearest-even
    return (unsigned short)(u >> 16);
}
__device__ __forceinline__ float bf16_f32(unsigned short h) {
    return __uint_as_float(((unsigned)h) << 16);
}
// pack {hi.bf16, lo.bf16} via one v_perm_b32 (truncation; bf16 has 8-bit
// mantissa so trunc-vs-RNE is immaterial here, accumulation is f32)
__device__ __forceinline__ unsigned pack2_bf16(float lo, float hi) {
    return __builtin_amdgcn_perm(__float_as_uint(hi), __float_as_uint(lo),
                                 0x07060302u);
}

// W1 [H x RH] fp32 -> bf16, laid out as [kchunk][n][32] so router blocks can
// stage a chunk with contiguous 128-bit copies.
__global__ void cvt_w1_kernel(const float* __restrict__ W1,
                              unsigned short* __restrict__ w1bf) {
    int i = blockIdx.x * blockDim.x + threadIdx.x;
    if (i >= H * RH) return;
    int k = i >> 9;          // / RH
    int n = i & (RH - 1);
    w1bf[(size_t)(k >> 5) * (RH * KC) + n * KC + (k & 31)] = f32_bf16(W1[i]);
}

__global__ __launch_bounds__(THREADS, 1)
void router_kernel(const float* __restrict__ X,             // [N,H]
                   const unsigned short* __restrict__ w1bf, // [NKC][RH][KC]
                   const float* __restrict__ b1,            // [RH]
                   const float* __restrict__ W2,            // [RH,E]
                   const float* __restrict__ b2,            // [E]
                   const float* __restrict__ scales,        // [E,H]
                   const float* __restrict__ biases,        // [E,H]
                   float* __restrict__ out,                 // [N,H]
                   int ntok) {
    __shared__ unsigned short ldsW[2 * WBUF];                 // 80 KB, double buffer
    __shared__ unsigned short ldsH[TOKS_PER_BLOCK * HSTRIDE]; // 65 KB hidden bf16
    __shared__ float ldsS[TOKS_PER_BLOCK * E];                // scores
    __shared__ int   ldsIdx[TOKS_PER_BLOCK];

    const int tid   = threadIdx.x;
    const int wave  = tid >> 5;
    const int lane  = tid & 31;
    const int tg    = wave >> 1;           // token group 0..3 (16 tokens each)
    const int nHalf = wave & 1;            // RH half 0..1 (256 cols each)
    const int laneM = lane & 15;           // M row within tile / N col within tile
    const int half  = lane >> 4;           // hi/lo K half per WMMA layouts

    const int tokBase = blockIdx.x * TOKS_PER_BLOCK + tg * 16;

    v8f acc[16];
    const v8f vzero = {0.f, 0.f, 0.f, 0.f, 0.f, 0.f, 0.f, 0.f};
#pragma unroll
    for (int t = 0; t < 16; ++t) acc[t] = vzero;

    int arowIdx = tokBase + laneM;
    if (arowIdx >= ntok) arowIdx = ntok - 1;   // safe clamp (N % 64 == 0 normally)
    const float* arow = X + (size_t)arowIdx * H;

    // staging index split: this thread copies uint4s [tid*8 .. tid*8+7],
    // uint4 idx -> column n = idx>>2, quarter q = idx&3
    const int sN = (tid * 8) >> 2;     // even thread mapping: 2 cols per thread
    // prologue: stage chunk 0 into buffer 0
    {
        const uint4* src = (const uint4*)(w1bf);
#pragma unroll
        for (int j = 0; j < 8; ++j) {
            int idx = tid * 8 + j;
            int n = idx >> 2, q = idx & 3;
            *((uint4*)(ldsW + n * WCOL_STRIDE) + q) = src[idx];
        }
    }
    __syncthreads();

    for (int kc = 0; kc < NKC; ++kc) {
        const int cur = kc & 1;
        const int nxt = cur ^ 1;

        // ---- issue next chunk's global loads first (latency hides under WMMA)
        uint4 stg[8];
        if (kc + 1 < NKC) {
            const uint4* src = (const uint4*)(w1bf + (size_t)(kc + 1) * (RH * KC));
#pragma unroll
            for (int j = 0; j < 8; ++j) stg[j] = src[tid * 8 + j];
            if (kc + 2 < NKC)
                __builtin_prefetch(w1bf + (size_t)(kc + 2) * (RH * KC), 0, 1);
        }

        // ---- A fragment: 16x32 bf16, contiguous K order (k = 16*half + s) ----
        // dword v holds K = half*16 + v*2 .. +1
        union { unsigned u[8]; v16bf v; } A;
        const float* ar = arow + kc * KC;
#pragma unroll
        for (int v = 0; v < 8; ++v) {
            int kb = (half << 4) + (v << 1);
            float2 f = *(const float2*)(ar + kb);
            A.u[v] = pack2_bf16(f.x, f.y);
        }

        // ---- 16 N-tiles: B frag from LDS (2x b128), then WMMA ----
        const unsigned short* wbuf = ldsW + cur * WBUF;
#pragma unroll
        for (int t = 0; t < 16; ++t) {
            int ncol = nHalf * 256 + t * 16 + laneM;   // lane's B column
            const uint4* bp = (const uint4*)(wbuf + ncol * WCOL_STRIDE + half * 16);
            union { uint4 q[2]; v16bf v; } B;
            B.q[0] = bp[0];
            B.q[1] = bp[1];
            acc[t] = __builtin_amdgcn_wmma_f32_16x16x32_bf16(
                false, A.v, false, B.v, (short)0, acc[t], false, false);
        }

        // ---- commit next chunk to the other buffer, single barrier per iter
        if (kc + 1 < NKC) {
            unsigned short* dbuf = ldsW + nxt * WBUF;
#pragma unroll
            for (int j = 0; j < 8; ++j) {
                int idx = tid * 8 + j;
                int n = idx >> 2, q = idx & 3;
                *((uint4*)(dbuf + n * WCOL_STRIDE) + q) = stg[j];
            }
        }
        __syncthreads();
    }

    // ---- bias + ReLU, write hidden (bf16) to LDS ----
    // C layout: vgpr r -> M = r + 8*half, N = laneM
#pragma unroll
    for (int t = 0; t < 16; ++t) {
        int ncol = nHalf * 256 + t * 16 + laneM;
        float bias = b1[ncol];
#pragma unroll
        for (int r = 0; r < 8; ++r) {
            float hv = acc[t][r] + bias;
            hv = hv > 0.f ? hv : 0.f;
            int m = (half << 3) + r;
            ldsH[(tg * 16 + m) * HSTRIDE + ncol] = f32_bf16(hv);
        }
    }
    // stage W2 (fp32, 32 KB) into the now-free ldsW region
    float* ldsW2 = (float*)ldsW;
    for (int i = tid; i < RH * E; i += THREADS) ldsW2[i] = W2[i];
    __syncthreads();

    // ---- GEMM2 + scores: 64 tokens x 16 experts, 4 pairs per thread ----
#pragma unroll
    for (int p = 0; p < 4; ++p) {
        int pid = tid * 4 + p;
        int tok = pid >> 4;
        int e   = pid & 15;
        float s = b2[e];
        const unsigned* hr32 = (const unsigned*)(ldsH + tok * HSTRIDE);
#pragma unroll 4
        for (int r2 = 0; r2 < RH / 2; ++r2) {
            unsigned hh = hr32[r2];
            s += bf16_f32((unsigned short)(hh & 0xFFFFu)) * ldsW2[(2 * r2) * E + e];
            s += bf16_f32((unsigned short)(hh >> 16))     * ldsW2[(2 * r2 + 1) * E + e];
        }
        ldsS[tok * E + e] = s;
    }
    __syncthreads();

    // ---- top-1 per token (first max, like jnp.argmax) ----
    if (tid < TOKS_PER_BLOCK) {
        const float* sr = ldsS + tid * E;
        float mx = sr[0];
        int mi = 0;
#pragma unroll
        for (int e = 1; e < E; ++e) {
            float v = sr[e];
            if (v > mx) { mx = v; mi = e; }
        }
        ldsIdx[tid] = mi;
    }
    __syncthreads();

    // ---- fused affine epilogue: out = X * scales[idx] + biases[idx] ----
    {
        int tok  = tid >> 2;            // 4 threads per token
        int l4   = tid & 3;
        int gtok = blockIdx.x * TOKS_PER_BLOCK + tok;
        if (gtok < ntok) {
            int idx = ldsIdx[tok];
            const float4* xr = (const float4*)(X + (size_t)gtok * H);
            const float4* sc = (const float4*)(scales + (size_t)idx * H);
            const float4* bi = (const float4*)(biases + (size_t)idx * H);
            float4* orow = (float4*)(out + (size_t)gtok * H);
            for (int h = l4; h < H / 4; h += 4) {
                float4 x = xr[h], s = sc[h], b = bi[h];
                float4 o;
                o.x = x.x * s.x + b.x;
                o.y = x.y * s.y + b.y;
                o.z = x.z * s.z + b.z;
                o.w = x.w * s.w + b.w;
                orow[h] = o;
            }
        }
    }
    (void)sN;
}

extern "C" void kernel_launch(void* const* d_in, const int* in_sizes, int n_in,
                              void* d_out, int out_size, void* d_ws, size_t ws_size,
                              hipStream_t stream) {
    const float* X      = (const float*)d_in[0];
    const float* W1     = (const float*)d_in[1];
    const float* b1     = (const float*)d_in[2];
    const float* W2     = (const float*)d_in[3];
    const float* b2     = (const float*)d_in[4];
    // d_in[5..8] = Wg1,bg1,Wg2,bg2: provably dead (gate > 0, softmax monotone)
    const float* scales = (const float*)d_in[9];
    const float* biases = (const float*)d_in[10];
    float* out = (float*)d_out;

    int ntok = in_sizes[0] / H;
    unsigned short* w1bf = (unsigned short*)d_ws;  // H*RH bf16 = 1 MB scratch

    cvt_w1_kernel<<<(H * RH + THREADS - 1) / THREADS, THREADS, 0, stream>>>(W1, w1bf);
    router_kernel<<<(ntok + TOKS_PER_BLOCK - 1) / TOKS_PER_BLOCK, THREADS, 0, stream>>>(
        X, w1bf, b1, W2, b2, scales, biases, out, ntok);
}
